// CrossTowerCausalModel_51273319579784
// MI455X (gfx1250) — compile-verified, matrix operand order checked by reading, not run on hardware
//
#include <hip/hip_runtime.h>
#include <cstdint>
#include <cstddef>

// ---------------------------------------------------------------------------
// Problem constants (match the reference)
// ---------------------------------------------------------------------------
constexpr int kH      = 768;
constexpr int kDSEM   = 1024;
constexpr int kNodes  = 32768;   // N = B*P
constexpr int kGraphs = 1024;    // B
constexpr int kP      = 32;      // nodes per graph
constexpr int kEPG    = 64;      // edges per graph
constexpr int kEdges  = 65536;   // E
constexpr int kLayers = 3;
constexpr int kH6     = 6 * kH;  // 4608

// ---------------------------------------------------------------------------
// Types for CDNA5 WMMA (wave32): v_wmma_f32_16x16x32_bf16
// ---------------------------------------------------------------------------
typedef __bf16 bf16;
typedef __attribute__((ext_vector_type(16))) __bf16      v16bf;
typedef __attribute__((ext_vector_type(4)))  __bf16      v4bf;
typedef __attribute__((ext_vector_type(8)))  float       v8f;
typedef __attribute__((ext_vector_type(4)))  unsigned    u32x4;

struct Frag { union { v16bf v; u32x4 q[2]; }; };

// A fragment (16x32 bf16, row-major source, leading dim = K):
//   lanes 0-15  hold row M=lane,    K = {0..7} (q[0]) and {16..23} (q[1])
//   lanes 16-31 hold row M=lane-16, K = {8..15}(q[0]) and {24..31} (q[1])
// B fragment mirrors this with lane = output column, loaded from the
// pre-transposed (N x K row-major) weight buffer — identical access pattern.
__device__ __forceinline__ void load_frag(Frag& f, const bf16* __restrict__ base,
                                          int ld, int row0, int k0, int lane) {
  const int m = lane & 15;
  const int g = (lane >> 4) << 3;                  // 0 or 8
  const bf16* p = base + (size_t)(row0 + m) * ld + (k0 + g);
  f.q[0] = *reinterpret_cast<const u32x4*>(p);        // K = k0+g    .. +8
  f.q[1] = *reinterpret_cast<const u32x4*>(p + 16);   // K = k0+16+g .. +8
}

__device__ __forceinline__ v8f wmma32(const Frag& a, const Frag& b, v8f c) {
  return __builtin_amdgcn_wmma_f32_16x16x32_bf16(false, a.v, false, b.v,
                                                 (short)0, c, false, false);
}

// One wave's 32x64 output region as a 2x4 grid of WMMA tiles.
// Per K-step: 6 fragments (12 b128 loads) feed 8 WMMAs -> 1.5 loads / wmma.
__device__ __forceinline__ void mma_k(const bf16* __restrict__ A,
                                      const bf16* __restrict__ Bt,
                                      int K, int r0, int c0, int lane,
                                      v8f acc[2][4]) {
  for (int k = 0; k < K; k += 32) {
    Frag a0, a1, b[4];
    load_frag(a0, A, K, r0,      k, lane);
    load_frag(a1, A, K, r0 + 16, k, lane);
#pragma unroll
    for (int j = 0; j < 4; ++j) load_frag(b[j], Bt, K, c0 + j * 16, k, lane);
    if (k + 32 < K) {  // global_prefetch_b8 for the streamed activation rows
      __builtin_prefetch(A + (size_t)(r0 + (lane & 15)) * K + k + 32, 0, 3);
    }
#pragma unroll
    for (int j = 0; j < 4; ++j) {
      acc[0][j] = wmma32(a0, b[j], acc[0][j]);
      acc[1][j] = wmma32(a1, b[j], acc[1][j]);
    }
  }
}

// C/D layout: VGPR r, lanes 0-15 -> (M = r,   N = lane)
//                     lanes16-31 -> (M = r+8, N = lane-16)

// Wave/block tiling: 256 threads = 8 waves in a 4(M) x 2(N) grid;
// block tile = 128 x 128; grid = (M/128, N/128), N = 768 -> grid.y = 6.

// ---------------------------------------------------------------------------
// GEMM 1: h_text = relu(x @ W_sem + spk_emb[sid] + emo_emb[eid])
// ---------------------------------------------------------------------------
__global__ __launch_bounds__(256) void gemm_htext_kernel(
    const bf16* __restrict__ A, const bf16* __restrict__ Bt,
    const int* __restrict__ sid, const int* __restrict__ eid,
    const float* __restrict__ spk, const float* __restrict__ emo,
    float* __restrict__ C) {
  const int lane = threadIdx.x & 31, wave = threadIdx.x >> 5;
  const int r0 = blockIdx.x * 128 + (wave >> 1) * 32;
  const int c0 = blockIdx.y * 128 + (wave & 1) * 64;
  v8f acc[2][4] = {};
  mma_k(A, Bt, kDSEM, r0, c0, lane, acc);
  const int g = (lane >> 4) << 3, n = lane & 15;
  for (int i = 0; i < 2; ++i)
    for (int j = 0; j < 4; ++j) {
      const int cb = c0 + j * 16 + n;
#pragma unroll
      for (int r = 0; r < 8; ++r) {
        const int row = r0 + i * 16 + g + r;
        float v = acc[i][j][r]
                + spk[(size_t)sid[row] * kH + cb]
                + emo[(size_t)eid[row] * kH + cb];
        C[(size_t)row * kH + cb] = v > 0.f ? v : 0.f;
      }
    }
}

// ---------------------------------------------------------------------------
// GNN layer: h_new = relu(h @ W_self + msg @ W_nbr) + h_old  (two fused GEMMs)
// ---------------------------------------------------------------------------
__global__ __launch_bounds__(256) void gemm_layer_kernel(
    const bf16* __restrict__ A1, const bf16* __restrict__ B1t,
    const bf16* __restrict__ A2, const bf16* __restrict__ B2t,
    const float* __restrict__ Hold, float* __restrict__ Hnew) {
  const int lane = threadIdx.x & 31, wave = threadIdx.x >> 5;
  const int r0 = blockIdx.x * 128 + (wave >> 1) * 32;
  const int c0 = blockIdx.y * 128 + (wave & 1) * 64;
  v8f acc[2][4] = {};
  mma_k(A1, B1t, kH, r0, c0, lane, acc);   // h   @ W_self
  mma_k(A2, B2t, kH, r0, c0, lane, acc);   // msg @ W_nbr (same accumulators)
  const int g = (lane >> 4) << 3, n = lane & 15;
  for (int i = 0; i < 2; ++i)
    for (int j = 0; j < 4; ++j) {
      const int cb = c0 + j * 16 + n;
#pragma unroll
      for (int r = 0; r < 8; ++r) {
        const int row = r0 + i * 16 + g + r;
        float v = acc[i][j][r];
        v = v > 0.f ? v : 0.f;
        Hnew[(size_t)row * kH + cb] = v + Hold[(size_t)row * kH + cb];
      }
    }
}

// ---------------------------------------------------------------------------
// Generic GEMM + bias + relu (z_teacher and W_p1 projections); N = 768
// ---------------------------------------------------------------------------
__global__ __launch_bounds__(256) void gemm_bias_relu_kernel(
    const bf16* __restrict__ A, const bf16* __restrict__ Bt,
    const float* __restrict__ bias, float* __restrict__ C, int K) {
  const int lane = threadIdx.x & 31, wave = threadIdx.x >> 5;
  const int r0 = blockIdx.x * 128 + (wave >> 1) * 32;
  const int c0 = blockIdx.y * 128 + (wave & 1) * 64;
  v8f acc[2][4] = {};
  mma_k(A, Bt, K, r0, c0, lane, acc);
  const int g = (lane >> 4) << 3, n = lane & 15;
  for (int i = 0; i < 2; ++i)
    for (int j = 0; j < 4; ++j) {
      const int cb = c0 + j * 16 + n;
      const float bb = bias[cb];
#pragma unroll
      for (int r = 0; r < 8; ++r) {
        const int row = r0 + i * 16 + g + r;
        float v = acc[i][j][r] + bb;
        C[(size_t)row * kH + cb] = v > 0.f ? v : 0.f;
      }
    }
}

// ---------------------------------------------------------------------------
// Pack / transpose helpers
// ---------------------------------------------------------------------------
__global__ void pack_bf16_kernel(const float* __restrict__ src,
                                 bf16* __restrict__ dst, long long n4) {
  long long i = (long long)blockIdx.x * blockDim.x + threadIdx.x;
  if (i >= n4) return;
  const float* p = src + i * 4;
  v4bf o = { (bf16)p[0], (bf16)p[1], (bf16)p[2], (bf16)p[3] };
  *reinterpret_cast<v4bf*>(dst + i * 4) = o;
}

// src: f32 K x N (row-major) -> dst: bf16 N x K (row-major)
__global__ __launch_bounds__(256) void transpose_pack_kernel(
    const float* __restrict__ src, bf16* __restrict__ dst, int K, int N) {
  __shared__ bf16 tile[32][33];
  const int kb = blockIdx.x * 32, nb = blockIdx.y * 32;
  const int tx = threadIdx.x & 31, ty = threadIdx.x >> 5;
  for (int i = ty; i < 32; i += 8) {
    const int k = kb + i, n = nb + tx;
    tile[i][tx] = (k < K && n < N) ? (bf16)src[(size_t)k * N + n] : (bf16)0.f;
  }
  __syncthreads();
  for (int i = ty; i < 32; i += 8) {
    const int n = nb + i, k = kb + tx;
    if (n < N && k < K) dst[(size_t)n * K + k] = tile[tx][i];
  }
}

// ---------------------------------------------------------------------------
// segment_sum per graph: msg[tgt] += h[src]; output directly as bf16.
// One block per graph; 16-node LDS accumulator (48 KB), feature-striped so
// each thread owns f == tid (mod 256) -> race-free without atomics/barriers.
// ---------------------------------------------------------------------------
__global__ __launch_bounds__(256) void msg_kernel(
    const float* __restrict__ h, const int* __restrict__ esrc,
    const int* __restrict__ etgt, bf16* __restrict__ msgb) {
  __shared__ float sm[16 * kH];
  const int g = blockIdx.x, tid = threadIdx.x;
  const int ebase = g * kEPG;
  for (int half = 0; half < 2; ++half) {
    for (int i = tid; i < 16 * kH; i += 256) sm[i] = 0.f;
    __syncthreads();
    for (int e = 0; e < kEPG; ++e) {
      const int s  = esrc[ebase + e];
      const int tl = etgt[ebase + e] - g * kP - half * 16;
      if (tl >= 0 && tl < 16) {
        const float* hr = h + (size_t)s * kH;
        for (int f = tid; f < kH; f += 256) sm[tl * kH + f] += hr[f];
      }
    }
    __syncthreads();
    for (int i = tid; i < 16 * kH; i += 256) {
      const int node = g * kP + half * 16 + i / kH;
      msgb[(size_t)node * kH + (i % kH)] = (bf16)sm[i];
    }
    __syncthreads();
  }
}

// ---------------------------------------------------------------------------
// first_edge / dist (replicates the reference exactly, incl. the local-index
// lookup into the global first_edge array)
// ---------------------------------------------------------------------------
__global__ void init_fe_kernel(int* fe, int n, int val) {
  int i = blockIdx.x * blockDim.x + threadIdx.x;
  if (i < n) fe[i] = val;
}
__global__ void fe_min_kernel(const int* __restrict__ esrc, int* fe, int E) {
  int e = blockIdx.x * blockDim.x + threadIdx.x;
  if (e < E) atomicMin(&fe[esrc[e]], e);
}
__global__ void dist_kernel(const int* __restrict__ tni, const int* __restrict__ fe,
                            const int* __restrict__ etgt, int* __restrict__ dist,
                            int B, int E) {
  int b = blockIdx.x * blockDim.x + threadIdx.x;
  if (b >= B) return;
  const int c = tni[2 * b], t = tni[2 * b + 1];
  const int fc = fe[c], ft = fe[t];
  const int cu = (fc < E) ? etgt[fc < E - 1 ? fc : E - 1] : c;
  const int tu = (ft < E) ? etgt[ft < E - 1 ? ft : E - 1] : t;
  int d = cu - tu; if (d < 0) d = -d; if (d > 31) d = 31;
  dist[b] = d;
}

// edge_repr = [h_c | h_text_c | h_t | h_text_t | dist_emb[d] | z], bf16
__global__ __launch_bounds__(256) void erepr_kernel(
    const float* __restrict__ hg, const float* __restrict__ ht,
    const float* __restrict__ demb, const float* __restrict__ z,
    const int* __restrict__ tni, const int* __restrict__ dist,
    bf16* __restrict__ er) {
  const int b = blockIdx.x;
  const int c = b * kP + tni[2 * b];
  const int t = b * kP + tni[2 * b + 1];
  const int d = dist[b];
  bf16* row = er + (size_t)b * kH6;
  for (int f = threadIdx.x; f < kH; f += blockDim.x) {
    row[f]          = (bf16)hg[(size_t)c * kH + f];
    row[kH + f]     = (bf16)ht[(size_t)c * kH + f];
    row[2 * kH + f] = (bf16)hg[(size_t)t * kH + f];
    row[3 * kH + f] = (bf16)ht[(size_t)t * kH + f];
    row[4 * kH + f] = (bf16)demb[(size_t)d * kH + f];
    row[5 * kH + f] = (bf16)z[(size_t)b * kH + f];
  }
}

// logits[b] = hid[b,:] . W_p2 + b_p2
__global__ __launch_bounds__(256) void logits_kernel(
    const float* __restrict__ hid, const float* __restrict__ W_p2,
    const float* __restrict__ b_p2, float* __restrict__ out) {
  __shared__ float red[256];
  const int b = blockIdx.x;
  float s = 0.f;
  for (int f = threadIdx.x; f < kH; f += 256)
    s += hid[(size_t)b * kH + f] * W_p2[f];
  red[threadIdx.x] = s;
  __syncthreads();
  for (int off = 128; off > 0; off >>= 1) {
    if (threadIdx.x < off) red[threadIdx.x] += red[threadIdx.x + off];
    __syncthreads();
  }
  if (threadIdx.x == 0) out[b] = red[0] + b_p2[0];
}

// ---------------------------------------------------------------------------
// Host orchestration
// ---------------------------------------------------------------------------
extern "C" void kernel_launch(void* const* d_in, const int* in_sizes, int n_in,
                              void* d_out, int out_size, void* d_ws, size_t ws_size,
                              hipStream_t stream) {
  (void)in_sizes; (void)n_in; (void)out_size; (void)ws_size;

  const float* x        = (const float*)d_in[0];
  const int*   sid      = (const int*)  d_in[1];
  const int*   eid      = (const int*)  d_in[2];
  const int*   eidx     = (const int*)  d_in[3];   // [2, E]
  const int*   tni      = (const int*)  d_in[4];   // [B, 2]
  const float* expl     = (const float*)d_in[5];
  const float* W_sem    = (const float*)d_in[6];
  const float* spk_emb  = (const float*)d_in[7];
  const float* emo_emb  = (const float*)d_in[8];
  const float* gw_self  = (const float*)d_in[9];
  const float* gw_nbr   = (const float*)d_in[10];
  const float* dist_emb = (const float*)d_in[11];
  const float* W_expl   = (const float*)d_in[12];
  const float* b_expl   = (const float*)d_in[13];
  const float* W_p1     = (const float*)d_in[14];
  const float* b_p1     = (const float*)d_in[15];
  const float* W_p2     = (const float*)d_in[16];
  const float* b_p2     = (const float*)d_in[17];

  const int* esrc = eidx;
  const int* etgt = eidx + kEdges;

  // workspace carve-up (256B aligned)
  size_t off = 0;
  auto take = [&](size_t bytes) -> void* {
    void* p = (char*)d_ws + off;
    off += (bytes + 255) & ~(size_t)255;
    return p;
  };
  bf16*  xb     = (bf16*) take((size_t)kNodes * kDSEM * 2);   // later reused as hbA
  bf16*  WsemT  = (bf16*) take((size_t)kH * kDSEM * 2);
  bf16*  WselfT = (bf16*) take((size_t)kLayers * kH * kH * 2);
  bf16*  WnbrT  = (bf16*) take((size_t)kLayers * kH * kH * 2);
  bf16*  WexplT = (bf16*) take((size_t)kH * kH * 2);
  bf16*  Wp1T   = (bf16*) take((size_t)kH * kH6 * 2);
  bf16*  explb  = (bf16*) take((size_t)kGraphs * kH * 2);
  float* htext  = (float*)take((size_t)kNodes * kH * 4);
  float* h_a    = (float*)take((size_t)kNodes * kH * 4);
  float* h_b    = (float*)take((size_t)kNodes * kH * 4);
  bf16*  hbB    = (bf16*) take((size_t)kNodes * kH * 2);
  bf16*  msgb   = (bf16*) take((size_t)kNodes * kH * 2);
  float* z      = (float*)take((size_t)kGraphs * kH * 4);
  int*   fe     = (int*)  take((size_t)kNodes * 4);
  int*   dist   = (int*)  take((size_t)kGraphs * 4);
  bf16*  erepr  = (bf16*) take((size_t)kGraphs * kH6 * 2);
  float* hid    = (float*)take((size_t)kGraphs * kH * 4);
  bf16*  hbA    = xb;  // xb is dead after GEMM 1; recycle as bf16 h buffer

  // 1) pack x and expl to bf16; transpose+pack all weights
  long long n4 = (long long)kNodes * kDSEM / 4;
  pack_bf16_kernel<<<(unsigned)((n4 + 255) / 256), 256, 0, stream>>>(x, xb, n4);
  n4 = (long long)kGraphs * kH / 4;
  pack_bf16_kernel<<<(unsigned)((n4 + 255) / 256), 256, 0, stream>>>(expl, explb, n4);

  transpose_pack_kernel<<<dim3(kDSEM / 32, kH / 32), 256, 0, stream>>>(W_sem, WsemT, kDSEM, kH);
  for (int l = 0; l < kLayers; ++l) {
    transpose_pack_kernel<<<dim3(kH / 32, kH / 32), 256, 0, stream>>>(
        gw_self + (size_t)l * kH * kH, WselfT + (size_t)l * kH * kH, kH, kH);
    transpose_pack_kernel<<<dim3(kH / 32, kH / 32), 256, 0, stream>>>(
        gw_nbr + (size_t)l * kH * kH, WnbrT + (size_t)l * kH * kH, kH, kH);
  }
  transpose_pack_kernel<<<dim3(kH / 32, kH / 32), 256, 0, stream>>>(W_expl, WexplT, kH, kH);
  transpose_pack_kernel<<<dim3(kH6 / 32, kH / 32), 256, 0, stream>>>(W_p1, Wp1T, kH6, kH);

  // 2) h_text = relu(x @ W_sem + spk + emo)
  gemm_htext_kernel<<<dim3(kNodes / 128, kH / 128), 256, 0, stream>>>(
      xb, WsemT, sid, eid, spk_emb, emo_emb, htext);

  // 3) GNN layers
  n4 = (long long)kNodes * kH / 4;
  pack_bf16_kernel<<<(unsigned)((n4 + 255) / 256), 256, 0, stream>>>(htext, hbA, n4);

  const float* hcur = htext;
  bf16* hbcur = hbA;
  float* houts[kLayers] = { h_a, h_b, h_a };
  for (int l = 0; l < kLayers; ++l) {
    msg_kernel<<<kGraphs, 256, 0, stream>>>(hcur, esrc, etgt, msgb);
    gemm_layer_kernel<<<dim3(kNodes / 128, kH / 128), 256, 0, stream>>>(
        hbcur, WselfT + (size_t)l * kH * kH,
        msgb,  WnbrT  + (size_t)l * kH * kH,
        hcur, houts[l]);
    hcur = houts[l];
    if (l + 1 < kLayers) {
      bf16* hbnext = (hbcur == hbA) ? hbB : hbA;
      pack_bf16_kernel<<<(unsigned)((n4 + 255) / 256), 256, 0, stream>>>(hcur, hbnext, n4);
      hbcur = hbnext;
    }
  }

  // 4) z_teacher = relu(expl @ W_expl + b_expl)
  gemm_bias_relu_kernel<<<dim3(kGraphs / 128, kH / 128), 256, 0, stream>>>(
      explb, WexplT, b_expl, z, kH);

  // 5) first_edge / dist
  init_fe_kernel<<<kNodes / 256, 256, 0, stream>>>(fe, kNodes, kEdges);
  fe_min_kernel<<<kEdges / 256, 256, 0, stream>>>(esrc, fe, kEdges);
  dist_kernel<<<kGraphs / 256, 256, 0, stream>>>(tni, fe, etgt, dist, kGraphs, kEdges);

  // 6) edge_repr gather (bf16) from final h, h_text, dist_emb, z
  erepr_kernel<<<kGraphs, 256, 0, stream>>>(hcur, htext, dist_emb, z, tni, dist, erepr);

  // 7) hid = relu(edge_repr @ W_p1 + b_p1); logits = hid @ W_p2 + b_p2
  gemm_bias_relu_kernel<<<dim3(kGraphs / 128, kH / 128), 256, 0, stream>>>(
      erepr, Wp1T, b_p1, hid, kH6);
  logits_kernel<<<kGraphs, 256, 0, stream>>>(hid, W_p2, b_p2, (float*)d_out);
}
